// U_model_32530082300017
// MI455X (gfx1250) — compile-verified, matrix operand order checked by reading, not run on hardware
//
#include <hip/hip_runtime.h>
#include <math.h>

// ---------------------------------------------------------------------------
// Model constants (from the reference)
// ---------------------------------------------------------------------------
#define NNODE 4096
#define NEDGE 131072
#define FDIM  128
#define VDIM  64
#define CUTI  (1.0f / 0.11f)
#define DIVBI (1.0f / 1.5625f)
#define AGGI  (1.0f / 16.0f)

typedef __attribute__((ext_vector_type(16))) __bf16 v16bf;
typedef __attribute__((ext_vector_type(8)))  float  v8f;

__device__ __forceinline__ unsigned short f2bf(float f) {
  unsigned int u = __float_as_uint(f);
  return (unsigned short)((u + 0x7FFFu + ((u >> 16) & 1u)) >> 16);  // RNE
}

union FragBF { uint4 u[2]; v16bf v; };

// ---------------------------------------------------------------------------
// CDNA5 async global->LDS copy (ASYNCcnt tracked, no VGPR round-trip).
// LDS operand is a byte offset from the wave's LDS base (dynamic LDS => 0).
// ---------------------------------------------------------------------------
__device__ __forceinline__ void async_ld128(unsigned int lds_byte_off, const void* gaddr) {
  asm volatile("global_load_async_to_lds_b128 %0, %1, off"
               :: "v"(lds_byte_off), "v"(gaddr) : "memory");
}
__device__ __forceinline__ void wait_async0() {
  asm volatile("s_wait_asynccnt 0x0" ::: "memory");
}

// ---------------------------------------------------------------------------
// Generic bf16 WMMA GEMM:  out = act(A[M][K](bf16) @ WT[N][K](bf16)^T + bias)
// Block: 256 thr = 8 waves; tile 128(M) x 64(N); K step 32.
// Double-buffered LDS, async global->LDS staging, one barrier per K step,
// all fragments loaded before the 4 back-to-back WMMAs.
// Requires: M%128==0, N%64==0, K%32==0 (K zero-padded).
// Dynamic LDS: 2*(128*32 + 64*32)*2 = 24576 bytes.
// ---------------------------------------------------------------------------
#define GEMM_LDS_BYTES 24576
#define A_ELEMS 4096   // 128*32 bf16 per A buffer
#define B_BASE  8192   // element offset of B buffers (2*A_ELEMS)
#define B_ELEMS 2048   // 64*32 bf16 per B buffer

__global__ __launch_bounds__(256) void gemm_bf16_kernel(
    const unsigned short* __restrict__ A, const unsigned short* __restrict__ WT,
    const float* __restrict__ bias, float* __restrict__ outF,
    unsigned short* __restrict__ outB, int M, int N, int K, int swish)
{
  extern __shared__ unsigned short lds[];
  const int tid  = threadIdx.x;
  const int wv   = tid >> 5;
  const int lane = tid & 31;
  const int hl   = lane >> 4;      // half-wave (K-interleave select)
  const int l15  = lane & 15;
  const int m0   = blockIdx.x * 128;
  const int n0   = blockIdx.y * 64;

  // per-thread staging chunks (16B each): A tile = 512 chunks (2/thread), B = 256
  const int rA0 = tid >> 2,          koA0 = (tid & 3) * 8;
  const int rA1 = (tid + 256) >> 2,  koA1 = (tid & 3) * 8;
  const int rB  = tid >> 2,          koB  = (tid & 3) * 8;

  auto issue = [&](int buf, int k0) {
    async_ld128((unsigned)((buf * A_ELEMS + rA0 * 32 + koA0) * 2),
                A + (size_t)(m0 + rA0) * K + k0 + koA0);
    async_ld128((unsigned)((buf * A_ELEMS + rA1 * 32 + koA1) * 2),
                A + (size_t)(m0 + rA1) * K + k0 + koA1);
    async_ld128((unsigned)((B_BASE + buf * B_ELEMS + rB * 32 + koB) * 2),
                WT + (size_t)(n0 + rB) * K + k0 + koB);
  };

  v8f zero = {0.f, 0.f, 0.f, 0.f, 0.f, 0.f, 0.f, 0.f};
  v8f acc[4];
#pragma unroll
  for (int j = 0; j < 4; ++j) acc[j] = zero;

  issue(0, 0);                       // prologue: tile 0 in flight
  int buf = 0;
  for (int k0 = 0; k0 < K; k0 += 32) {
    wait_async0();                   // this thread's async stores to LDS done
    __syncthreads();                 // tile k0 visible; prev reads of other buf done
    if (k0 + 32 < K) issue(buf ^ 1, k0 + 32);   // overlap next tile with WMMAs

    const unsigned short* Ab = lds + buf * A_ELEMS;
    const unsigned short* Bb = lds + B_BASE + buf * B_ELEMS;

    // load ALL fragments first (one dscnt wait), then stream 4 WMMAs
    FragBF a, b[4];
    a.u[0] = *(const uint4*)&Ab[(wv * 16 + l15) * 32 + hl * 8];
    a.u[1] = *(const uint4*)&Ab[(wv * 16 + l15) * 32 + hl * 8 + 16];
#pragma unroll
    for (int j = 0; j < 4; ++j) {
      b[j].u[0] = *(const uint4*)&Bb[(j * 16 + l15) * 32 + hl * 8];
      b[j].u[1] = *(const uint4*)&Bb[(j * 16 + l15) * 32 + hl * 8 + 16];
    }
#pragma unroll
    for (int j = 0; j < 4; ++j) {
      acc[j] = __builtin_amdgcn_wmma_f32_16x16x32_bf16(
          false, a.v, false, b[j].v, (short)0, acc[j], false, false);
    }
    buf ^= 1;
  }

  // epilogue: bias + swish, write f32 and/or bf16
#pragma unroll
  for (int j = 0; j < 4; ++j) {
    int col = n0 + j * 16 + l15;
    float bv = bias ? bias[col] : 0.0f;
#pragma unroll
    for (int i = 0; i < 8; ++i) {
      int row = m0 + wv * 16 + hl * 8 + i;
      float v = acc[j][i] + bv;
      if (swish) v = v / (1.0f + expf(-v));
      if (outF) outF[(size_t)row * N + col] = v;
      if (outB) outB[(size_t)row * N + col] = f2bf(v);
    }
  }
}

// ---------------------------------------------------------------------------
// Weight prep: W[K][N] f32 -> WT[N][Kpad] bf16 (zero-padded K)
// ---------------------------------------------------------------------------
__global__ void prep_w_kernel(const float* __restrict__ W,
                              unsigned short* __restrict__ WT,
                              int K, int N, int Kpad)
{
  size_t idx = (size_t)blockIdx.x * 256 + threadIdx.x;
  if (idx >= (size_t)N * Kpad) return;
  int k = (int)(idx % Kpad);
  int n = (int)(idx / Kpad);
  WT[idx] = (k < K) ? f2bf(W[(size_t)k * N + n]) : (unsigned short)0;
}

// ---------------------------------------------------------------------------
// Edge graph construction (reproduces jnp.where row-major order)
// ---------------------------------------------------------------------------
__device__ __forceinline__ float pair_d2(const float* __restrict__ x, int i, int j,
                                         float xi0, float xi1, float xi2,
                                         float* dr)
{
  float d0 = xi0 - x[j * 3 + 0]; d0 -= rintf(d0); d0 *= CUTI;
  float d1 = xi1 - x[j * 3 + 1]; d1 -= rintf(d1); d1 *= CUTI;
  float d2 = xi2 - x[j * 3 + 2]; d2 -= rintf(d2); d2 *= CUTI;
  float s = d0 * d0 + d1 * d1 + d2 * d2;
  if (i == j) s += 10.0f;
  if (j == 0) s *= DIVBI;
  if (i == 0) s *= DIVBI;
  if (dr) { dr[0] = d0; dr[1] = d1; dr[2] = d2; }
  return s;
}

__global__ __launch_bounds__(256) void edge_count_kernel(const float* __restrict__ x,
                                                         int* __restrict__ rowcnt)
{
  int i = blockIdx.x;
  float xi0 = x[i * 3], xi1 = x[i * 3 + 1], xi2 = x[i * 3 + 2];
  int cnt = 0;
  for (int j = threadIdx.x; j < NNODE; j += 256)
    cnt += (pair_d2(x, i, j, xi0, xi1, xi2, nullptr) < 1.0f) ? 1 : 0;
  __shared__ int sm[256];
  sm[threadIdx.x] = cnt; __syncthreads();
  for (int s = 128; s > 0; s >>= 1) {
    if (threadIdx.x < s) sm[threadIdx.x] += sm[threadIdx.x + s];
    __syncthreads();
  }
  if (threadIdx.x == 0) rowcnt[i] = sm[0];
}

__global__ void row_scan_kernel(const int* __restrict__ rowcnt, int* __restrict__ rowoff)
{
  if (blockIdx.x == 0 && threadIdx.x == 0) {
    int acc = 0;
    for (int i = 0; i < NNODE; ++i) { rowoff[i] = acc; acc += rowcnt[i]; }
    rowoff[NNODE] = acc;
  }
}

__global__ void edge_default_kernel(int* snd, int* rcv, float* d2e, float* maske, float* edR)
{
  int e = blockIdx.x * 256 + threadIdx.x;
  if (e >= NEDGE) return;
  snd[e] = -42; rcv[e] = -42; d2e[e] = 0.f; maske[e] = 0.f;
  edR[(size_t)e * 3] = 0.f; edR[(size_t)e * 3 + 1] = 0.f; edR[(size_t)e * 3 + 2] = 0.f;
}

__global__ __launch_bounds__(256) void edge_write_kernel(
    const float* __restrict__ x, const int* __restrict__ rowoff,
    int* snd, int* rcv, float* d2e, float* maske, float* edR)
{
  int i = blockIdx.x;
  int wv = threadIdx.x >> 5, lane = threadIdx.x & 31;
  __shared__ int wsum[8];
  __shared__ int base_s;
  if (threadIdx.x == 0) base_s = rowoff[i];
  float xi0 = x[i * 3], xi1 = x[i * 3 + 1], xi2 = x[i * 3 + 2];
  __syncthreads();
  for (int j0 = 0; j0 < NNODE; j0 += 256) {
    int j = j0 + threadIdx.x;
    float dr[3];
    float d2 = pair_d2(x, i, j, xi0, xi1, xi2, dr);
    bool p = d2 < 1.0f;
    unsigned long long bm = __ballot(p);
    int lpre = __popcll(bm & ((1ull << lane) - 1ull));
    if (lane == 0) wsum[wv] = __popcll(bm);
    __syncthreads();
    int woff = 0, tot = 0;
    for (int w = 0; w < 8; ++w) { if (w < wv) woff += wsum[w]; tot += wsum[w]; }
    if (p) {
      int idx = base_s + woff + lpre;
      if (idx < NEDGE) {
        snd[idx] = i; rcv[idx] = j; d2e[idx] = d2; maske[idx] = 1.0f;
        edR[(size_t)idx * 3] = dr[0]; edR[(size_t)idx * 3 + 1] = dr[1]; edR[(size_t)idx * 3 + 2] = dr[2];
      }
    }
    __syncthreads();
    if (threadIdx.x == 0) base_s += tot;
    __syncthreads();
  }
}

// ---------------------------------------------------------------------------
// Node / edge-state initialization
// ---------------------------------------------------------------------------
__global__ void node_h0_kernel(const float* __restrict__ x, float* __restrict__ h0)
{
  int i = blockIdx.x * 256 + threadIdx.x;
  if (i >= NNODE) return;
  float d = pair_d2(x, i, 0, x[i * 3], x[i * 3 + 1], x[i * 3 + 2], nullptr);
  h0[i * 2 + 0] = (i == 0) ? 1.0f : 0.0f;
  h0[i * 2 + 1] = d;
}

__global__ void node_embed_kernel(const float* __restrict__ h0, const float* __restrict__ t,
                                  const float* __restrict__ W, float* __restrict__ hN)
{
  int idx = blockIdx.x * 256 + threadIdx.x;
  if (idx >= NNODE * FDIM) return;
  int i = idx / FDIM, f = idx % FDIM;
  hN[idx] = h0[i * 2] * W[f] + h0[i * 2 + 1] * W[FDIM + f] + t[0] * W[2 * FDIM + f];
}

// hEV[e][x][v] = edR[e][x] * Wemb[v]     (h_edge_vec layout: [E][3][64])
__global__ void hev_init_kernel(const float* __restrict__ edR,
                                const float* __restrict__ Wemb, float* __restrict__ hEV)
{
  size_t idx = (size_t)blockIdx.x * 256 + threadIdx.x;
  if (idx >= (size_t)NEDGE * 3 * VDIM) return;
  int v = (int)(idx % VDIM);
  size_t ex = idx / VDIM;
  hEV[idx] = edR[ex] * Wemb[v];
}

// edge_mlp0 input: [d2, h0[s](2), h0[r](2), t] padded to 32, bf16
__global__ void pack_em0_kernel(const int* __restrict__ snd, const int* __restrict__ rcv,
                                const float* __restrict__ d2e, const float* __restrict__ h0,
                                const float* __restrict__ t, unsigned short* __restrict__ out)
{
  size_t idx = (size_t)blockIdx.x * 256 + threadIdx.x;
  if (idx >= (size_t)NEDGE * 32) return;
  int c = (int)(idx % 32);
  int e = (int)(idx / 32);
  int s = snd[e] & (NNODE - 1), r = rcv[e] & (NNODE - 1);
  float v = 0.f;
  switch (c) {
    case 0: v = d2e[e]; break;
    case 1: v = h0[s * 2]; break;
    case 2: v = h0[s * 2 + 1]; break;
    case 3: v = h0[r * 2]; break;
    case 4: v = h0[r * 2 + 1]; break;
    case 5: v = t[0]; break;
    default: v = 0.f;
  }
  out[idx] = f2bf(v);
}

// ---------------------------------------------------------------------------
// Per-layer packing / scatter / elementwise
// ---------------------------------------------------------------------------
__device__ __forceinline__ float dot3(const float* a, const float* b, int v) {
  return a[v] * b[v] + a[64 + v] * b[64 + v] + a[128 + v] * b[128 + v];
}

// msg input row (769 -> pad 800) in bf16
__global__ __launch_bounds__(256) void pack_msg_kernel(
    const int* __restrict__ snd, const int* __restrict__ rcv,
    const float* __restrict__ hN, const float* __restrict__ hV,
    const float* __restrict__ hE, const float* __restrict__ hEV,
    const float* __restrict__ t, unsigned short* __restrict__ out)
{
  int e = blockIdx.x;
  int s = snd[e] & (NNODE - 1), r = rcv[e] & (NNODE - 1);
  __shared__ float svs[192], svr[192], sev[192];
  for (int k = threadIdx.x; k < 192; k += 256) {
    svs[k] = hV[(size_t)s * 192 + k];
    svr[k] = hV[(size_t)r * 192 + k];
    sev[k] = hEV[(size_t)e * 192 + k];
  }
  __syncthreads();
  unsigned short* row = out + (size_t)e * 800;
  for (int c = threadIdx.x; c < 800; c += 256) {
    float v;
    if      (c <  64) v = dot3(svr, sev, c);
    else if (c < 128) v = dot3(svs, sev, c - 64);
    else if (c < 192) v = dot3(svs, svr, c - 128);
    else if (c < 256) v = dot3(svs, svs, c - 192);
    else if (c < 320) v = dot3(svr, svr, c - 256);
    else if (c < 384) v = dot3(sev, sev, c - 320);
    else if (c < 512) v = hN[(size_t)s * 128 + (c - 384)];
    else if (c < 640) v = hN[(size_t)r * 128 + (c - 512)];
    else if (c < 768) v = hE[(size_t)e * 128 + (c - 640)];
    else if (c == 768) v = t[0];
    else v = 0.0f;
    row[c] = f2bf(v);
  }
}

// masked cosine-cutoff scatter-add into node aggregates
__global__ __launch_bounds__(256) void scatter_kernel(
    const int* __restrict__ snd, const int* __restrict__ rcv,
    const float* __restrict__ mw_all, const float* __restrict__ d2e,
    const float* __restrict__ maske, const float* __restrict__ hN,
    const float* __restrict__ hEV, float* __restrict__ h_agg,
    float* __restrict__ hv_agg)
{
  int e = blockIdx.x;
  int s = snd[e] & (NNODE - 1), r = rcv[e] & (NNODE - 1);
  float cwm = 0.5f * (cosf(d2e[e] * 3.14159265358979323846f) + 1.0f) * maske[e];
  const float* mw = mw_all + (size_t)e * 192;
  for (int f = threadIdx.x; f < 128; f += 256)
    atomicAdd(&h_agg[(size_t)r * 128 + f], mw[f] * cwm * hN[(size_t)s * 128 + f]);
  for (int c = threadIdx.x; c < 192; c += 256) {
    int v = c & 63;
    atomicAdd(&hv_agg[(size_t)r * 192 + c], hEV[(size_t)e * 192 + c] * mw[128 + v] * cwm);
  }
}

__global__ void cvt_bf16_kernel(const float* __restrict__ in, unsigned short* __restrict__ out,
                                size_t n, float scale)
{
  size_t i = (size_t)blockIdx.x * 256 + threadIdx.x;
  if (i < n) out[i] = f2bf(in[i] * scale);
}

// hev rows (e,x): [hEV[e][x][:], hvL[s][x][:], hvL[r][x][:]] -> bf16 [E*3][192]
__global__ void pack_hev_kernel(const int* __restrict__ snd, const int* __restrict__ rcv,
                                const float* __restrict__ hEV, const float* __restrict__ hvL,
                                unsigned short* __restrict__ out)
{
  size_t idx = (size_t)blockIdx.x * 256 + threadIdx.x;
  if (idx >= (size_t)NEDGE * 3 * 192) return;
  int c = (int)(idx % 192);
  size_t ex = idx / 192;
  int xcoord = (int)(ex % 3);
  int e = (int)(ex / 3);
  float v;
  if (c < 64) v = hEV[ex * 64 + c];
  else if (c < 128) { int s = snd[e] & (NNODE - 1); v = hvL[((size_t)s * 3 + xcoord) * 64 + (c - 64)]; }
  else              { int r = rcv[e] & (NNODE - 1); v = hvL[((size_t)r * 3 + xcoord) * 64 + (c - 128)]; }
  out[idx] = f2bf(v);
}

// edge_mlp input: [hE(128), hL[s](128), hL[r](128)] -> bf16 [E][384]
__global__ void pack_edgein_kernel(const int* __restrict__ snd, const int* __restrict__ rcv,
                                   const float* __restrict__ hE, const float* __restrict__ hL,
                                   unsigned short* __restrict__ out)
{
  size_t idx = (size_t)blockIdx.x * 256 + threadIdx.x;
  if (idx >= (size_t)NEDGE * 384) return;
  int c = (int)(idx % 384);
  int e = (int)(idx / 384);
  float v;
  if (c < 128) v = hE[(size_t)e * 128 + c];
  else if (c < 256) { int s = snd[e] & (NNODE - 1); v = hL[(size_t)s * 128 + (c - 128)]; }
  else              { int r = rcv[e] & (NNODE - 1); v = hL[(size_t)r * 128 + (c - 256)]; }
  out[idx] = f2bf(v);
}

__global__ void add_kernel(float* __restrict__ a, const float* __restrict__ b, size_t n)
{
  size_t i = (size_t)blockIdx.x * 256 + threadIdx.x;
  if (i < n) a[i] += b[i];
}

__global__ __launch_bounds__(256) void reduce_mean_kernel(const float* __restrict__ h,
                                                          float* __restrict__ out)
{
  __shared__ float sm[256];
  const size_t n = (size_t)NNODE * FDIM;
  float acc = 0.f;
  for (size_t i = (size_t)blockIdx.x * 256 + threadIdx.x; i < n; i += (size_t)gridDim.x * 256)
    acc += h[i];
  sm[threadIdx.x] = acc; __syncthreads();
  for (int s = 128; s > 0; s >>= 1) {
    if (threadIdx.x < s) sm[threadIdx.x] += sm[threadIdx.x + s];
    __syncthreads();
  }
  if (threadIdx.x == 0) atomicAdd(out, sm[0] * (1.0f / ((float)NNODE * (float)FDIM)));
}

// ---------------------------------------------------------------------------
// Host orchestration
// Param leaf order (alphabetical pytree flatten):
//   0 edge_embed_W | 1..6 edge_mlp0 (W,b x3) | per layer l (base 7+20l):
//   +0..5 edge_mlp, +6 edge_vec_W, +7 hvec_W, +8..13 msg, +14..19 node_mlp
//   87 node_embed_W | 88 t | 89 x
// ---------------------------------------------------------------------------
extern "C" void kernel_launch(void* const* d_in, const int* in_sizes, int n_in,
                              void* d_out, int out_size, void* d_ws, size_t ws_size,
                              hipStream_t stream)
{
  (void)in_sizes; (void)n_in; (void)out_size; (void)ws_size;
  auto P = [&](int i) { return (const float*)d_in[i]; };
  const float* t = P(88);
  const float* x = P(89);

  size_t off = 0;
  auto carve = [&](size_t bytes) -> void* {
    void* p = (char*)d_ws + off;
    off += (bytes + 255) & ~(size_t)255;
    return p;
  };
  auto prep = [&](const float* W, int K, int N, int Kpad) -> unsigned short* {
    unsigned short* wt = (unsigned short*)carve((size_t)N * Kpad * 2);
    size_t tot = (size_t)N * Kpad;
    prep_w_kernel<<<dim3((unsigned)((tot + 255) / 256)), dim3(256), 0, stream>>>(W, wt, K, N, Kpad);
    return wt;
  };
  auto gemm = [&](const unsigned short* A, const unsigned short* WT, const float* bias,
                  float* oF, unsigned short* oB, int M, int N, int K, int swish) {
    gemm_bf16_kernel<<<dim3((unsigned)(M / 128), (unsigned)(N / 64)), dim3(256),
                       GEMM_LDS_BYTES, stream>>>(A, WT, bias, oF, oB, M, N, K, swish);
  };

  // ---- weight prep (f32 -> transposed, padded bf16) ----
  unsigned short* em0_w0 = prep(P(1), 6, 256, 32);
  unsigned short* em0_w1 = prep(P(3), 256, 256, 256);
  unsigned short* em0_w2 = prep(P(5), 256, 128, 256);
  unsigned short *msg_w0[4], *msg_w1[4], *msg_w2[4];
  unsigned short *nod_w0[4], *nod_w1[4], *nod_w2[4];
  unsigned short *edg_w0[4], *edg_w1[4], *edg_w2[4];
  unsigned short *hvw[4], *evw[4];
  for (int l = 0; l < 4; ++l) {
    int b = 7 + 20 * l;
    edg_w0[l] = prep(P(b + 0), 384, 256, 384);
    edg_w1[l] = prep(P(b + 2), 256, 256, 256);
    edg_w2[l] = prep(P(b + 4), 256, 128, 256);
    evw[l]    = prep(P(b + 6), 192, 64, 192);
    hvw[l]    = prep(P(b + 7), 64, 64, 64);
    msg_w0[l] = prep(P(b + 8), 769, 256, 800);
    msg_w1[l] = prep(P(b + 10), 256, 256, 256);
    msg_w2[l] = prep(P(b + 12), 256, 192, 256);
    nod_w0[l] = prep(P(b + 14), 128, 256, 128);
    nod_w1[l] = prep(P(b + 16), 256, 256, 256);
    nod_w2[l] = prep(P(b + 18), 256, 128, 256);
  }

  // ---- activation / state buffers ----
  int*   snd    = (int*)carve((size_t)NEDGE * 4);
  int*   rcv    = (int*)carve((size_t)NEDGE * 4);
  float* d2e    = (float*)carve((size_t)NEDGE * 4);
  float* maske  = (float*)carve((size_t)NEDGE * 4);
  float* edR    = (float*)carve((size_t)NEDGE * 3 * 4);
  int*   rowcnt = (int*)carve((size_t)NNODE * 4);
  int*   rowoff = (int*)carve((size_t)(NNODE + 1) * 4);
  float* h0     = (float*)carve((size_t)NNODE * 2 * 4);
  float* hN     = (float*)carve((size_t)NNODE * 128 * 4);
  float* hV     = (float*)carve((size_t)NNODE * 192 * 4);
  float* hE     = (float*)carve((size_t)NEDGE * 128 * 4);
  float* hEV    = (float*)carve((size_t)NEDGE * 192 * 4);
  unsigned short* em0in  = (unsigned short*)carve((size_t)NEDGE * 32 * 2);
  unsigned short* msgin  = (unsigned short*)carve((size_t)NEDGE * 800 * 2);
  unsigned short* actA   = (unsigned short*)carve((size_t)NEDGE * 256 * 2);
  unsigned short* actB   = (unsigned short*)carve((size_t)NEDGE * 256 * 2);
  float* mw_all = (float*)carve((size_t)NEDGE * 192 * 4);
  float* h_agg  = (float*)carve((size_t)NNODE * 128 * 4);
  float* hv_agg = (float*)carve((size_t)NNODE * 192 * 4);
  unsigned short* nbf = (unsigned short*)carve((size_t)NNODE * 192 * 2);
  float* hL     = (float*)carve((size_t)NNODE * 128 * 4);
  float* hvL    = (float*)carve((size_t)NNODE * 192 * 4);
  unsigned short* hevbf  = (unsigned short*)carve((size_t)NEDGE * 3 * 192 * 2);
  float* hevL   = (float*)carve((size_t)NEDGE * 192 * 4);
  unsigned short* edgein = (unsigned short*)carve((size_t)NEDGE * 384 * 2);
  float* heL    = (float*)carve((size_t)NEDGE * 128 * 4);

  // ---- graph build ----
  edge_default_kernel<<<dim3(NEDGE / 256), dim3(256), 0, stream>>>(snd, rcv, d2e, maske, edR);
  edge_count_kernel<<<dim3(NNODE), dim3(256), 0, stream>>>(x, rowcnt);
  row_scan_kernel<<<dim3(1), dim3(1), 0, stream>>>(rowcnt, rowoff);
  edge_write_kernel<<<dim3(NNODE), dim3(256), 0, stream>>>(x, rowoff, snd, rcv, d2e, maske, edR);

  // ---- initial node / edge embeddings ----
  node_h0_kernel<<<dim3(NNODE / 256), dim3(256), 0, stream>>>(x, h0);
  node_embed_kernel<<<dim3((NNODE * FDIM) / 256), dim3(256), 0, stream>>>(h0, t, P(87), hN);
  hipMemsetAsync(hV, 0, (size_t)NNODE * 192 * 4, stream);
  hev_init_kernel<<<dim3((unsigned)(((size_t)NEDGE * 3 * VDIM) / 256)), dim3(256), 0, stream>>>(
      edR, P(0), hEV);
  pack_em0_kernel<<<dim3((unsigned)(((size_t)NEDGE * 32) / 256)), dim3(256), 0, stream>>>(
      snd, rcv, d2e, h0, t, em0in);
  gemm(em0in, em0_w0, P(2), nullptr, actA, NEDGE, 256, 32, 1);
  gemm(actA,  em0_w1, P(4), nullptr, actB, NEDGE, 256, 256, 1);
  gemm(actB,  em0_w2, P(6), hE, nullptr, NEDGE, 128, 256, 0);

  // ---- message-passing layers ----
  for (int l = 0; l < 4; ++l) {
    int b = 7 + 20 * l;
    // msg MLP (769 -> 256 -> 256 -> 192)
    pack_msg_kernel<<<dim3(NEDGE), dim3(256), 0, stream>>>(snd, rcv, hN, hV, hE, hEV, t, msgin);
    gemm(msgin, msg_w0[l], P(b + 9),  nullptr, actA, NEDGE, 256, 800, 1);
    gemm(actA,  msg_w1[l], P(b + 11), nullptr, actB, NEDGE, 256, 256, 1);
    gemm(actB,  msg_w2[l], P(b + 13), mw_all, nullptr, NEDGE, 192, 256, 0);
    // aggregation
    hipMemsetAsync(h_agg, 0, (size_t)NNODE * 128 * 4, stream);
    hipMemsetAsync(hv_agg, 0, (size_t)NNODE * 192 * 4, stream);
    scatter_kernel<<<dim3(NEDGE), dim3(256), 0, stream>>>(snd, rcv, mw_all, d2e, maske,
                                                          hN, hEV, h_agg, hv_agg);
    // node MLP on h_agg/16
    cvt_bf16_kernel<<<dim3((NNODE * 128) / 256), dim3(256), 0, stream>>>(
        h_agg, nbf, (size_t)NNODE * 128, AGGI);
    gemm(nbf,  nod_w0[l], P(b + 15), nullptr, actA, NNODE, 256, 128, 1);
    gemm(actA, nod_w1[l], P(b + 17), nullptr, actB, NNODE, 256, 256, 1);
    gemm(actB, nod_w2[l], P(b + 19), hL, nullptr, NNODE, 128, 256, 0);
    // h_vec einsum (per (n,x): 64x64 GEMM)
    cvt_bf16_kernel<<<dim3((NNODE * 192) / 256), dim3(256), 0, stream>>>(
        hv_agg, nbf, (size_t)NNODE * 192, AGGI);
    gemm(nbf, hvw[l], nullptr, hvL, nullptr, NNODE * 3, 64, 64, 0);
    // edge_vec einsum (per (e,x): K=192 -> 64)
    pack_hev_kernel<<<dim3((unsigned)(((size_t)NEDGE * 3 * 192) / 256)), dim3(256), 0, stream>>>(
        snd, rcv, hEV, hvL, hevbf);
    gemm(hevbf, evw[l], nullptr, hevL, nullptr, NEDGE * 3, 64, 192, 0);
    // edge MLP (384 -> 256 -> 256 -> 128)
    pack_edgein_kernel<<<dim3((unsigned)(((size_t)NEDGE * 384) / 256)), dim3(256), 0, stream>>>(
        snd, rcv, hE, hL, edgein);
    gemm(edgein, edg_w0[l], P(b + 1), nullptr, actA, NEDGE, 256, 384, 1);
    gemm(actA,   edg_w1[l], P(b + 3), nullptr, actB, NEDGE, 256, 256, 1);
    gemm(actB,   edg_w2[l], P(b + 5), heL, nullptr, NEDGE, 128, 256, 0);
    // residual updates
    add_kernel<<<dim3((NNODE * 128) / 256), dim3(256), 0, stream>>>(hN, hL, (size_t)NNODE * 128);
    add_kernel<<<dim3((NNODE * 192) / 256), dim3(256), 0, stream>>>(hV, hvL, (size_t)NNODE * 192);
    add_kernel<<<dim3((unsigned)(((size_t)NEDGE * 128) / 256)), dim3(256), 0, stream>>>(
        hE, heL, (size_t)NEDGE * 128);
    add_kernel<<<dim3((unsigned)(((size_t)NEDGE * 192) / 256)), dim3(256), 0, stream>>>(
        hEV, hevL, (size_t)NEDGE * 192);
  }

  // ---- output: mean(h) ----
  hipMemsetAsync(d_out, 0, sizeof(float), stream);
  reduce_mean_kernel<<<dim3(256), dim3(256), 0, stream>>>(hN, (float*)d_out);
}
